// Module_45200235823830
// MI455X (gfx1250) — matrix operand order repaired and checked by reading, not compiled, not run on hardware
//
#include <hip/hip_runtime.h>
#include <math.h>

// ---------------- constants (match reference) ----------------
#define N_ITEMS   50000
#define Dk        128
#define Lh        200
#define LP        208          // L padded to 13 tiles of 16
#define NTILES    13
#define TAU_INV   0.25f
#define SIGMA_C   0.1f
#define PRIOR_C   1.0f
#define EPS_C     1e-8f
#define LN_EPS_C  1e-5f

// ---------------- dynamic LDS layout (bytes, 16B aligned) ----------------
#define OFF_AST    0          // 2 x 16x128 f32  double-buffered async gather staging
#define OFF_HID    16384      // 208x128 f16     h_id (V matrix + GEMM2 A)
#define OFF_KWT    69632      // 16x128  f32     KW tile
#define OFF_SCORE  77824      // 2x208   f32     scores (u,t)
#define OFF_ATTN   79488      // 2x208   f32     attn weights
#define OFF_MASK   81152      // 208     f32     maskf
#define OFF_HIDX   81984      // 208     i32     hist indices
#define OFF_UID    82816      // 128     f32     u_id
#define OFF_TRAW   83328      // 128     f32     t_raw
#define OFF_TID    83840      // 128     f32     t_id
#define OFF_QWU    84352      // 128     f32     u_id @ Wq^T
#define OFF_QWT    84864      // 128     f32     t_id @ Wq^T
#define OFF_OUTV   85376      // 2x128   f32     attention outputs
#define OFF_UVEC   86400      // 2x128   f32     layernormed u,i
#define OFF_VAT    87424      // 128     f32     v_attn
#define OFF_RED    87936      // 256     f32     reduction scratch
#define OFF_SACC   88960      // 32      f32     per-tile score accum
#define SMEM_BYTES 89088

typedef __attribute__((ext_vector_type(16))) _Float16 v16h;
typedef __attribute__((ext_vector_type(8)))  float    v8f;

// A fragment (16x32) from an LDS f32 tile (row stride Dk), converting to f16.
// lanes 0-15: row=lane, K = kb+{0..7, 16..23}; lanes 16-31: row=lane-16, K = kb+{8..15, 24..31}
__device__ __forceinline__ v16h load_A_f32_lds(const float* base, int lane, int kb) {
  const float* r = base + (size_t)(lane & 15) * Dk + kb + ((lane & 16) ? 8 : 0);
  v16h a;
#pragma unroll
  for (int j = 0; j < 8; ++j) { a[j] = (_Float16)r[j]; a[8 + j] = (_Float16)r[16 + j]; }
  return a;
}

// A fragment (16x32 f16) from an LDS f16 tile with row stride Dk.
__device__ __forceinline__ v16h load_A_lds(const _Float16* base, int lane, int kb) {
  const _Float16* r = base + (size_t)(lane & 15) * Dk + kb + ((lane & 16) ? 8 : 0);
  v16h a;
#pragma unroll
  for (int j = 0; j < 8; ++j) { a[j] = r[j]; a[8 + j] = r[16 + j]; }
  return a;
}

// B fragment (32x16 f16) for B = W^T directly from global f32 W (row-major, stride Dk).
// lanes 0-15: col n=lane, K = kb+{0..7, 16..23}; lanes 16-31: col n=lane-16, K = kb+{8..15, 24..31}
__device__ __forceinline__ v16h load_Bt_global(const float* W, int lane, int kb, int nbase) {
  const float* r = W + (size_t)(nbase + (lane & 15)) * Dk + kb + ((lane & 16) ? 8 : 0);
  v16h b;
#pragma unroll
  for (int j = 0; j < 8; ++j) { b[j] = (_Float16)r[j]; b[8 + j] = (_Float16)r[16 + j]; }
  return b;
}

__device__ __forceinline__ float blk_sum(float v, float* red, int tid) {
  red[tid] = v; __syncthreads();
#pragma unroll
  for (int s = 128; s > 0; s >>= 1) { if (tid < s) red[tid] += red[tid + s]; __syncthreads(); }
  float r = red[0]; __syncthreads(); return r;
}

__device__ __forceinline__ float blk_max(float v, float* red, int tid) {
  red[tid] = v; __syncthreads();
#pragma unroll
  for (int s = 128; s > 0; s >>= 1) { if (tid < s) red[tid] = fmaxf(red[tid], red[tid + s]); __syncthreads(); }
  float r = red[0]; __syncthreads(); return r;
}

__global__ __launch_bounds__(256)
void bam_forward_kernel(const int* __restrict__ user_idx, const int* __restrict__ item_idx,
                        const int* __restrict__ user_hist,
                        const float* __restrict__ user_embed, const float* __restrict__ item_embed,
                        const float* __restrict__ W_i, const float* __restrict__ W_h,
                        const float* __restrict__ Wq, const float* __restrict__ Wk,
                        const float* __restrict__ v_attn, const float* __restrict__ pred_W,
                        const float* __restrict__ pred_b,
                        const float* __restrict__ gamma_u, const float* __restrict__ beta_u,
                        const float* __restrict__ gamma_i, const float* __restrict__ beta_i,
                        const float* __restrict__ noise_u, const float* __restrict__ noise_i,
                        float* __restrict__ logits, float* __restrict__ kl_ws) {
  extern __shared__ __align__(16) char smem[];
  float*    ast   = (float*)(smem + OFF_AST);      // [2][16][128] f32
  _Float16* hid   = (_Float16*)(smem + OFF_HID);
  float*    kwt   = (float*)(smem + OFF_KWT);
  float*    score = (float*)(smem + OFF_SCORE);
  float*    attn  = (float*)(smem + OFF_ATTN);
  float*    maskf = (float*)(smem + OFF_MASK);
  int*      hidx  = (int*)(smem + OFF_HIDX);
  float*    s_uid = (float*)(smem + OFF_UID);
  float*    s_trw = (float*)(smem + OFF_TRAW);
  float*    s_tid = (float*)(smem + OFF_TID);
  float*    s_qwu = (float*)(smem + OFF_QWU);
  float*    s_qwt = (float*)(smem + OFF_QWT);
  float*    outv  = (float*)(smem + OFF_OUTV);
  float*    uvec  = (float*)(smem + OFF_UVEC);
  float*    s_vat = (float*)(smem + OFF_VAT);
  float*    red   = (float*)(smem + OFF_RED);
  float*    sacc  = (float*)(smem + OFF_SACC);

  const int b   = blockIdx.x;
  const int tid = threadIdx.x;
  const int uix = user_idx[b];
  const int iix = item_idx[b];

  // ---- Phase 0: stage vectors / hist / mask into LDS ----
  if (tid < Dk) {
    s_uid[tid] = user_embed[(size_t)uix * Dk + tid];
    s_trw[tid] = item_embed[(size_t)iix * Dk + tid];
    s_vat[tid] = v_attn[tid];
  }
  for (int l = tid; l < LP; l += 256) {
    int hl = (l < Lh) ? user_hist[(size_t)uix * Lh + l] : N_ITEMS;
    hidx[l]  = hl;
    maskf[l] = (hl != iix && hl != N_ITEMS) ? 1.0f : 0.0f;
  }
  __syncthreads();

  // t_id = t_raw @ W_i^T ; then qWu = u_id @ Wq^T, qWt = t_id @ Wq^T
  if (tid < Dk) {
    float acc = 0.0f;
    const float* wr = W_i + (size_t)tid * Dk;
#pragma unroll 4
    for (int k = 0; k < Dk; ++k) acc = fmaf(wr[k], s_trw[k], acc);
    s_tid[tid] = acc;
  }
  __syncthreads();
  if (tid < Dk) {
    float a0 = 0.0f, a1 = 0.0f;
    const float* wr = Wq + (size_t)tid * Dk;
#pragma unroll 4
    for (int k = 0; k < Dk; ++k) { float w = wr[k]; a0 = fmaf(w, s_uid[k], a0); a1 = fmaf(w, s_tid[k], a1); }
    s_qwu[tid] = a0; s_qwt[tid] = a1;
  }

  const int wave  = tid >> 5;
  const int lane  = tid & 31;
  const int nbase = wave * 16;                 // each wave owns 16 output columns
  const int c_m   = (lane & 16) ? 8 : 0;       // C fragment row base for this lane half
  const int c_n   = nbase + (lane & 15);       // C fragment column for this lane

  // Loop-invariant B fragments held in registers (W_h and Wk columns of this wave).
  v16h Bh[4], Bk2[4];
#pragma unroll
  for (int kt = 0; kt < 4; ++kt) {
    Bh[kt]  = load_Bt_global(W_h, lane, kt * 32, nbase);
    Bk2[kt] = load_Bt_global(Wk,  lane, kt * 32, nbase);
  }

  // Async gather of one 16x128 f32 tile into staging buffer `buf`.
  // 512 chunks of 16B; each of 256 threads issues exactly 2 async b128 copies.
  auto issue_tile = [&](int t, int buf) {
#pragma unroll
    for (int c = tid; c < 512; c += 256) {
      int row = c >> 5;
      unsigned boff = (unsigned)(c & 31) << 4;
      int hl = hidx[t * 16 + row];
      const char* g = (const char*)(item_embed + (size_t)hl * Dk) + boff;
      unsigned lds = (unsigned)(OFF_AST + buf * 8192 + (row << 9)) + boff;
      asm volatile("global_load_async_to_lds_b128 %0, %1, off"
                   :: "v"(lds), "v"(g) : "memory");
    }
  };

  __syncthreads();          // hidx / qW vectors visible
  issue_tile(0, 0);         // prologue: stage tile 0

  // ---- Phase 1: per-16-row tile: async gather -> GEMM1 (h_id) -> GEMM2 (KW) -> scores ----
  for (int t0 = 0; t0 < NTILES; ++t0) {
    const int buf = t0 & 1;
    if (tid < 32) sacc[tid] = 0.0f;
    if (t0 + 1 < NTILES) {
      issue_tile(t0 + 1, buf ^ 1);                       // prefetch next tile
      asm volatile("s_wait_asynccnt 0x2" ::: "memory");  // wait only for current tile
    } else {
      asm volatile("s_wait_asynccnt 0x0" ::: "memory");
    }
    __syncthreads();   // current tile staged + sacc zeroed, visible to all waves

    // GEMM1: h_id tile cols [nbase, nbase+16) = ast(16x128 f32) @ W_h^T
    const float* atile32 = ast + buf * 2048;
    v8f c1 = {};
#pragma unroll
    for (int kt = 0; kt < 4; ++kt) {
      v16h a = load_A_f32_lds(atile32, lane, kt * 32);
      c1 = __builtin_amdgcn_wmma_f32_16x16x32_f16(false, a, false, Bh[kt], (short)0, c1, false, false);
    }
    {
      _Float16* hrow = hid + (size_t)(t0 * 16 + c_m) * Dk + c_n;
#pragma unroll
      for (int j = 0; j < 8; ++j) hrow[(size_t)j * Dk] = (_Float16)c1[j];
    }
    __syncthreads();   // full h_id tile visible

    // GEMM2: KW tile cols [nbase, nbase+16) = h_id_tile(16x128 f16) @ Wk^T
    v8f c2 = {};
    const _Float16* atile = hid + (size_t)t0 * 16 * Dk;
#pragma unroll
    for (int kt = 0; kt < 4; ++kt) {
      v16h a = load_A_lds(atile, lane, kt * 32);
      c2 = __builtin_amdgcn_wmma_f32_16x16x32_f16(false, a, false, Bk2[kt], (short)0, c2, false, false);
    }
    {
      float* krow = kwt + (size_t)c_m * Dk + c_n;
#pragma unroll
      for (int j = 0; j < 8; ++j) krow[(size_t)j * Dk] = c2[j];
    }
    __syncthreads();

    // scores: s[q][row] = sum_n v[n] * tanh(qW[q][n] + KW[row][n])
    {
      int q   = tid >> 7;           // 0 = user query, 1 = target query
      int idx = tid & 127;
      int row = idx >> 3;
      int cb  = (idx & 7) * 16;
      const float* qW = q ? s_qwt : s_qwu;
      float p = 0.0f;
#pragma unroll
      for (int j = 0; j < 16; ++j) {
        int cc = cb + j;
        p = fmaf(s_vat[cc], tanhf(qW[cc] + kwt[row * Dk + cc]), p);
      }
      atomicAdd(&sacc[q * 16 + row], p);
    }
    __syncthreads();
    if (tid < 32) {
      int q = tid >> 4, r = tid & 15, l = t0 * 16 + r;
      score[q * LP + l] = (maskf[l] > 0.0f) ? sacc[tid] * TAU_INV : -1e9f;
    }
    __syncthreads();
  }

  // ---- Phase 2: masked softmax -> lognormal sample -> attn + KL (per query) ----
  float kl_num[2];
  float kl_den = 0.0f;
  const float mf = (tid < LP) ? maskf[tid] : 0.0f;
  for (int q = 0; q < 2; ++q) {
    float sc = (tid < LP) ? score[q * LP + tid] : -1e30f;
    float mx = blk_max(sc, red, tid);
    float ex = (tid < LP) ? expf(sc - mx) : 0.0f;
    float se = blk_sum(ex, red, tid);
    float phi = ex / se;
    float mu  = logf(phi + EPS_C);
    float nz  = 0.0f;
    if (tid < Lh) nz = q ? noise_i[(size_t)b * Lh + tid] : noise_u[(size_t)b * Lh + tid];
    float wgt = expf(mu + SIGMA_C * nz) * mf;
    float wsm = blk_sum(wgt, red, tid);
    if (tid < LP) attn[q * LP + tid] = wgt / (wsm + EPS_C);
    float klel = (logf(PRIOR_C / SIGMA_C) + (SIGMA_C * SIGMA_C + mu * mu) * (0.5f / (PRIOR_C * PRIOR_C)) - 0.5f) * mf;
    kl_num[q] = blk_sum(klel, red, tid);
    if (q == 0) kl_den = blk_sum(mf, red, tid);
  }
  __syncthreads();

  // ---- Phase 3: out[q][d] = sum_l attn[q][l] * h_id[l][d] ----
  {
    int q = tid >> 7, d = tid & 127;
    const float* aq = attn + q * LP;
    float acc = 0.0f;
#pragma unroll 4
    for (int l = 0; l < LP; ++l) acc = fmaf(aq[l], (float)hid[(size_t)l * Dk + d], acc);
    outv[q * Dk + d] = acc;
  }
  __syncthreads();

  // ---- Phase 4: layernorm(u_hist*u_id), layernorm(i_hist*t_id), logit ----
  for (int q = 0; q < 2; ++q) {
    float x = 0.0f;
    if (tid < Dk) x = outv[q * Dk + tid] * (q ? s_tid[tid] : s_uid[tid]);
    float mean = blk_sum(x, red, tid) * (1.0f / Dk);
    float dx = (tid < Dk) ? (x - mean) : 0.0f;
    float var = blk_sum(dx * dx, red, tid) * (1.0f / Dk);
    if (tid < Dk) {
      float g  = q ? gamma_i[tid] : gamma_u[tid];
      float be = q ? beta_i[tid]  : beta_u[tid];
      uvec[q * Dk + tid] = dx * rsqrtf(var + LN_EPS_C) * g + be;
    }
    __syncthreads();
  }
  float pl = (tid < Dk) ? uvec[tid] * uvec[Dk + tid] * pred_W[tid] : 0.0f;
  float lg = blk_sum(pl, red, tid);
  if (tid == 0) {
    logits[b] = lg + pred_b[0];
    kl_ws[(size_t)b * 3 + 0] = kl_num[0];
    kl_ws[(size_t)b * 3 + 1] = kl_num[1];
    kl_ws[(size_t)b * 3 + 2] = kl_den;
  }
}

__global__ __launch_bounds__(256)
void kl_reduce_kernel(const float* __restrict__ ws, float* __restrict__ out, int Bsz) {
  __shared__ float red[256];
  int tid = threadIdx.x;
  float n0 = 0.0f, n1 = 0.0f, dd = 0.0f;
  for (int i = tid; i < Bsz; i += 256) {
    const float* p = ws + (size_t)i * 3;
    n0 += p[0]; n1 += p[1]; dd += p[2];
  }
  float N0 = blk_sum(n0, red, tid);
  float N1 = blk_sum(n1, red, tid);
  float DD = blk_sum(dd, red, tid);
  if (tid == 0) {
    float inv = 1.0f / (DD + EPS_C);
    out[Bsz] = 0.5f * (N0 * inv + N1 * inv);
  }
}

extern "C" void kernel_launch(void* const* d_in, const int* in_sizes, int n_in,
                              void* d_out, int out_size, void* d_ws, size_t ws_size,
                              hipStream_t stream) {
  const int*   user_idx   = (const int*)  d_in[0];
  const int*   item_idx   = (const int*)  d_in[1];
  const int*   user_hist  = (const int*)  d_in[2];
  const float* user_embed = (const float*)d_in[3];
  const float* item_embed = (const float*)d_in[4];
  const float* W_i        = (const float*)d_in[5];
  const float* W_h        = (const float*)d_in[6];
  const float* Wq         = (const float*)d_in[7];
  const float* Wk         = (const float*)d_in[8];
  const float* v_attn     = (const float*)d_in[9];
  const float* pred_W     = (const float*)d_in[10];
  const float* pred_b     = (const float*)d_in[11];
  const float* gamma_u    = (const float*)d_in[12];
  const float* beta_u     = (const float*)d_in[13];
  const float* gamma_i    = (const float*)d_in[14];
  const float* beta_i     = (const float*)d_in[15];
  const float* noise_u    = (const float*)d_in[16];
  const float* noise_i    = (const float*)d_in[17];

  const int Bsz = in_sizes[0];
  float* logits = (float*)d_out;        // d_out[0..B-1] = logits, d_out[B] = KL scalar
  float* kl_ws  = (float*)d_ws;         // 3 floats per batch element

  bam_forward_kernel<<<dim3(Bsz), dim3(256), SMEM_BYTES, stream>>>(
      user_idx, item_idx, user_hist, user_embed, item_embed, W_i, W_h, Wq, Wk,
      v_attn, pred_W, pred_b, gamma_u, beta_u, gamma_i, beta_i, noise_u, noise_i,
      logits, kl_ws);
  kl_reduce_kernel<<<dim3(1), dim3(256), 0, stream>>>((const float*)d_ws, (float*)d_out, Bsz);
}
